// AdaptiveURBFLayer_34462817583499
// MI455X (gfx1250) — compile-verified, matrix operand order checked by reading, not run on hardware
//
#include <hip/hip_runtime.h>
#include <cstdint>
#include <cstddef>

#define BATCH 8192
#define IN_F  256
#define OUT_F 4096

// ---- WMMA vector types (gfx1250, wave32) ----
typedef __attribute__((ext_vector_type(16))) __bf16 v16bf;
typedef __attribute__((ext_vector_type(8)))  float  v8f;

union TileU {
  v16bf v;
  uint4 q[2];
};

// f32 -> bf16 round-to-nearest-even, done in integer so no scalar __bf16 math
// is needed on the host-compile pass.
__device__ __forceinline__ uint16_t f32_to_bf16_rne(float f) {
  uint32_t b = __float_as_uint(f);
  uint32_t r = b + 0x7FFFu + ((b >> 16) & 1u);
  return (uint16_t)(r >> 16);
}
__device__ __forceinline__ float bf16_bits_to_f32(uint16_t h) {
  return __uint_as_float(((uint32_t)h) << 16);
}

// Load one 16x32 (rows x K) bf16 operand tile in the ISA 7.12.2 layout.
// Works for both A (rows = batch rows of x) and B (rows = rows of E, i.e.
// columns of W = E^T, which are contiguous in K).
//   lanes 0-15 : row = lane,    halves 0..7 = K+0..7,  halves 8..15 = K+16..23
//   lanes 16-31: row = lane-16, halves 0..7 = K+8..15, halves 8..15 = K+24..31
__device__ __forceinline__ v16bf load_tile16x32(const uint16_t* base, int ld,
                                                int row0, int kb, int lane) {
  const int r    = lane & 15;
  const int koff = (lane >> 4) << 3;           // 0 or 8
  const uint16_t* p = base + (size_t)(row0 + r) * ld + (kb + koff);
  TileU t;
  t.q[0] = *(const uint4*)(p);                 // 8 bf16 = 16 B
  t.q[1] = *(const uint4*)(p + 16);            // K += 16
  return t.v;
}

// ---- prep: x(f32) -> x_hi,x_lo (bf16 split), E(f32) -> bf16 ----
__global__ __launch_bounds__(256)
void urbf_prep(const float* __restrict__ x, const float* __restrict__ e,
               uint16_t* __restrict__ xhi, uint16_t* __restrict__ xlo,
               uint16_t* __restrict__ eb) {
  const int i = blockIdx.x * blockDim.x + threadIdx.x;
  if (i < BATCH * IN_F) {
    float v = x[i];
    uint16_t h = f32_to_bf16_rne(v);
    xhi[i] = h;
    xlo[i] = f32_to_bf16_rne(v - bf16_bits_to_f32(h));   // exact residual, then RNE
  }
  if (i < OUT_F * IN_F) {
    eb[i] = f32_to_bf16_rne(e[i]);                       // E is {0,1}: exact
  }
}

// ---- main GEMM + fused RBF epilogue ----
// Block = 256 threads = 8 waves; wave (w>>2) picks M half-tile, (w&3) picks a
// 64-wide N strip. Block tile = 32(M) x 256(N). Grid = (8192/32, 4096/256).
__global__ __launch_bounds__(256)
void urbf_wmma(const uint16_t* __restrict__ xhi, const uint16_t* __restrict__ xlo,
               const uint16_t* __restrict__ eb,
               const float* __restrict__ means, const float* __restrict__ vars,
               const float* __restrict__ coefs, float* __restrict__ out) {
  const int lane = threadIdx.x & 31;
  const int wave = threadIdx.x >> 5;
  const int m0 = blockIdx.x * 32  + (wave >> 2) * 16;
  const int n0 = blockIdx.y * 256 + (wave & 3) * 64;

  v8f acc[4] = {};

  for (int kb = 0; kb < IN_F; kb += 32) {
    v16bf ahi = load_tile16x32(xhi, IN_F, m0, kb, lane);
    v16bf alo = load_tile16x32(xlo, IN_F, m0, kb, lane);

    // Prefetch next K-step of A (speculative; harmless at the tail).
    __builtin_prefetch(xhi + (size_t)(m0 + (lane & 15)) * IN_F + kb + 32, 0, 3);

#pragma unroll
    for (int t = 0; t < 4; ++t) {
      v16bf b = load_tile16x32(eb, IN_F, n0 + t * 16, kb, lane);
      // hi and lo parts accumulate into the same f32 accumulator:
      // xe = (x_hi + x_lo) @ E^T  ->  ~f32 accuracy from bf16 WMMA.
      acc[t] = __builtin_amdgcn_wmma_f32_16x16x32_bf16(
          false, ahi, false, b, (short)0, acc[t], false, false);
      acc[t] = __builtin_amdgcn_wmma_f32_16x16x32_bf16(
          false, alo, false, b, (short)0, acc[t], false, false);
    }
  }

  // C/D layout: VGPR r, lane L -> M = r + (L>=16)*8, N = L&15.
  const int nl   = lane & 15;
  const int moff = (lane >> 4) << 3;
#pragma unroll
  for (int t = 0; t < 4; ++t) {
    const int n = n0 + t * 16 + nl;
    const float mu = means[n];
    const float rv = 1.0f / vars[n];
    const float cf = coefs[n];
#pragma unroll
    for (int r = 0; r < 8; ++r) {
      const float z = (acc[t][r] - mu) * rv;
      out[(size_t)(m0 + moff + r) * OUT_F + n] = __expf(-0.5f * z * z) * cf;
    }
  }
}

// ---- generic fallback (only if workspace is too small): dense dot ----
__global__ __launch_bounds__(256)
void urbf_naive(const float* __restrict__ x, const float* __restrict__ e,
                const float* __restrict__ means, const float* __restrict__ vars,
                const float* __restrict__ coefs, float* __restrict__ out) {
  const size_t i = (size_t)blockIdx.x * blockDim.x + threadIdx.x;
  if (i >= (size_t)BATCH * OUT_F) return;
  const int    n = (int)(i & (OUT_F - 1));
  const size_t b = i >> 12;                 // / OUT_F
  const float* xr = x + b * IN_F;
  const float* er = e + (size_t)n * IN_F;
  float s = 0.f;
  for (int k = 0; k < IN_F; ++k) s += xr[k] * er[k];
  const float z = (s - means[n]) / vars[n];
  out[i] = __expf(-0.5f * z * z) * coefs[n];
}

extern "C" void kernel_launch(void* const* d_in, const int* in_sizes, int n_in,
                              void* d_out, int out_size, void* d_ws, size_t ws_size,
                              hipStream_t stream) {
  (void)in_sizes; (void)n_in; (void)out_size;
  const float* x     = (const float*)d_in[0];
  const float* e     = (const float*)d_in[1];
  const float* means = (const float*)d_in[2];
  const float* vars_ = (const float*)d_in[3];
  const float* coefs = (const float*)d_in[4];
  float* out = (float*)d_out;

  const size_t nx   = (size_t)BATCH * IN_F;   // 2 Mi elements
  const size_t ne   = (size_t)OUT_F * IN_F;   // 1 Mi elements
  const size_t need = nx * 2u * sizeof(uint16_t) + ne * sizeof(uint16_t); // 10 MiB

  if (ws_size >= need) {
    uint16_t* xhi = (uint16_t*)d_ws;
    uint16_t* xlo = xhi + nx;
    uint16_t* eb  = xlo + nx;

    urbf_prep<<<(unsigned)((nx + 255) / 256), 256, 0, stream>>>(x, e, xhi, xlo, eb);

    dim3 grid(BATCH / 32, OUT_F / 256);
    urbf_wmma<<<grid, 256, 0, stream>>>(xhi, xlo, eb, means, vars_, coefs, out);
  } else {
    const size_t total = (size_t)BATCH * OUT_F;
    urbf_naive<<<(unsigned)((total + 255) / 256), 256, 0, stream>>>(
        x, e, means, vars_, coefs, out);
  }
}
